// OutputLayer_8272107012447
// MI455X (gfx1250) — compile-verified, matrix operand order checked by reading, not run on hardware
//
#include <hip/hip_runtime.h>
#include <hip/hip_bf16.h>
#include <math.h>

// Problem sizes (fixed by the reference): B=8, S=2048, H=4096
#define MDIM 16384   // B*S rows
#define NDIM 4096    // H outputs
#define KDIM 4096    // H inputs
#define BM 128
#define BN 128

typedef __attribute__((ext_vector_type(16))) __bf16 v16bf;
typedef __attribute__((ext_vector_type(8)))  float  v8f;

union FragBF {
    v16bf v;
    uint4 q[2];
};

__device__ __forceinline__ float gelu_tanh(float h) {
    const float c0 = 0.7978845608028654f; // sqrt(2/pi)
    float t = tanhf(c0 * (h + 0.044715f * h * h * h));
    return 0.5f * h * (1.0f + t);
}

// ---------------- CDNA5 async-copy helpers ----------------
// LDS address is taken from the VDST VGPR plus the instruction offset (ISA 15.18); the low 32 bits of a
// flat pointer to __shared__ are the workgroup LDS byte offset.
__device__ __forceinline__ void async_copy_b128(unsigned lds_off, const void* gptr) {
    asm volatile("global_load_async_to_lds_b128 %0, %1, off"
                 :: "v"(lds_off), "v"((unsigned long long)(uintptr_t)gptr)
                 : "memory");
}
#define WAIT_ASYNC(n) asm volatile("s_wait_asynccnt %0" :: "i"(n) : "memory")

// =====================================================================
// Path A: bf16 inputs pre-converted into workspace, async-to-LDS GEMM
// =====================================================================
#define ABK 64        // K per LDS tile (2 WMMA K-steps)
#define ALDT 72       // padded row stride in bf16: 36 DWORDs -> conflict-free b128

__global__ __launch_bounds__(256)
void cvt_f32_to_bf16_kernel(const float* __restrict__ in, __bf16* __restrict__ out) {
    size_t i = (size_t)blockIdx.x * blockDim.x + threadIdx.x; // one float4 each
    float4 f = ((const float4*)in)[i];
    union { __bf16 h[4]; uint2 u; } p;
    p.h[0] = (__bf16)f.x; p.h[1] = (__bf16)f.y;
    p.h[2] = (__bf16)f.z; p.h[3] = (__bf16)f.w;
    ((uint2*)out)[i] = p.u;
}

__global__ __launch_bounds__(256)
void gemm_async_kernel(const __bf16* __restrict__ Xb,
                       const __bf16* __restrict__ Wb,
                       const float* __restrict__ bias,
                       float* __restrict__ Out)
{
    __shared__ __bf16 As[2][BM * ALDT];   // 18432 B per buffer
    __shared__ __bf16 Bs[2][BN * ALDT];   // total 73728 B

    const int tid  = threadIdx.x;
    const int lane = tid & 31;
    const int wave = tid >> 5;
    const int r16  = lane & 15;
    const int kh   = lane >> 4;
    const int wm   = wave & 3;    // M offset wm*32
    const int wn   = wave >> 2;   // N offset wn*64

    const int blockM = blockIdx.y * BM;
    const int blockN = blockIdx.x * BN;

    // Per-thread async-copy slots: p = tid + i*256, row = p>>3, c8 = p&7
    // (tile = 128 rows x 64 bf16 = 8 x 16B chunks per row)
    const __bf16* ga[4];
    const __bf16* gb[4];
    unsigned la[4], lb[4];
#pragma unroll
    for (int i = 0; i < 4; ++i) {
        int p   = tid + i * 256;
        int row = p >> 3;
        int c8  = p & 7;
        ga[i] = Xb + (size_t)(blockM + row) * KDIM + c8 * 8;
        gb[i] = Wb + (size_t)(blockN + row) * KDIM + c8 * 8;
        la[i] = (unsigned)(uintptr_t)&As[0][row * ALDT + c8 * 8];
        lb[i] = (unsigned)(uintptr_t)&Bs[0][row * ALDT + c8 * 8];
    }
    const unsigned bufA = (unsigned)((uintptr_t)&As[1][0] - (uintptr_t)&As[0][0]);
    const unsigned bufB = (unsigned)((uintptr_t)&Bs[1][0] - (uintptr_t)&Bs[0][0]);

    v8f acc[2][4];
#pragma unroll
    for (int mi = 0; mi < 2; ++mi)
#pragma unroll
        for (int ni = 0; ni < 4; ++ni)
            acc[mi][ni] = v8f{};

    auto issueAsync = [&](int buf) {
#pragma unroll
        for (int i = 0; i < 4; ++i) {
            async_copy_b128(la[i] + buf * bufA, ga[i]);
            async_copy_b128(lb[i] + buf * bufB, gb[i]);
            ga[i] += ABK;               // advance K for next tile
            gb[i] += ABK;
        }
    };

    const int KT = KDIM / ABK;   // 64 tiles

    issueAsync(0);
    for (int kt = 0; kt < KT; ++kt) {
        const int cur = kt & 1;
        if (kt + 1 < KT) {
            issueAsync(cur ^ 1);
            WAIT_ASYNC(8);           // tile kt complete (async loads retire in order)
        } else {
            WAIT_ASYNC(0);
        }
        __syncthreads();             // tile kt visible from all waves

#pragma unroll
        for (int ks = 0; ks < 2; ++ks) {       // two WMMA K-steps per tile
            FragBF a[2], b[4];
#pragma unroll
            for (int mi = 0; mi < 2; ++mi) {
                int row = wm * 32 + mi * 16 + r16;
                const __bf16* base = &As[cur][row * ALDT + ks * 32 + kh * 8];
                a[mi].q[0] = *(const uint4*)(base);
                a[mi].q[1] = *(const uint4*)(base + 16);
            }
#pragma unroll
            for (int ni = 0; ni < 4; ++ni) {
                int row = wn * 64 + ni * 16 + r16;
                const __bf16* base = &Bs[cur][row * ALDT + ks * 32 + kh * 8];
                b[ni].q[0] = *(const uint4*)(base);
                b[ni].q[1] = *(const uint4*)(base + 16);
            }
#pragma unroll
            for (int mi = 0; mi < 2; ++mi)
#pragma unroll
                for (int ni = 0; ni < 4; ++ni)
                    acc[mi][ni] = __builtin_amdgcn_wmma_f32_16x16x32_bf16(
                        false, a[mi].v, false, b[ni].v,
                        (short)0, acc[mi][ni], false, false);
        }
        __syncthreads();             // all reads of buf cur done before overwrite
    }

#pragma unroll
    for (int mi = 0; mi < 2; ++mi) {
#pragma unroll
        for (int ni = 0; ni < 4; ++ni) {
            int n = blockN + wn * 64 + ni * 16 + r16;
            float bv = bias[n];
#pragma unroll
            for (int r = 0; r < 8; ++r) {
                int m = blockM + wm * 32 + mi * 16 + kh * 8 + r;
                float h = acc[mi][ni][r] + bv;
                Out[(size_t)m * NDIM + n] = gelu_tanh(h);
            }
        }
    }
}

// =====================================================================
// Path B (fallback, ws too small): fused f32->bf16 staging GEMM (round-1)
// =====================================================================
#define FBK 32
#define FLDT 40

__global__ __launch_bounds__(256)
void gemm_fused_kernel(const float* __restrict__ X,
                       const float* __restrict__ W,
                       const float* __restrict__ bias,
                       float* __restrict__ Out)
{
    __shared__ __bf16 As[2][BM * FLDT];
    __shared__ __bf16 Bs[2][BN * FLDT];

    const int tid  = threadIdx.x;
    const int lane = tid & 31;
    const int wave = tid >> 5;
    const int r16  = lane & 15;
    const int kh   = lane >> 4;
    const int wm   = wave & 3;
    const int wn   = wave >> 2;

    const int blockM = blockIdx.y * BM;
    const int blockN = blockIdx.x * BN;

    float4 ra[4], rb[4];
    v8f acc[2][4];
#pragma unroll
    for (int mi = 0; mi < 2; ++mi)
#pragma unroll
        for (int ni = 0; ni < 4; ++ni)
            acc[mi][ni] = v8f{};

    auto loadRegs = [&](int kt) {
        const float* ax = X + (size_t)blockM * KDIM + kt * FBK;
        const float* bx = W + (size_t)blockN * KDIM + kt * FBK;
#pragma unroll
        for (int i = 0; i < 4; ++i) {
            int p = tid + i * 256, row = p >> 3, c4 = p & 7;
            ra[i] = *(const float4*)(ax + (size_t)row * KDIM + c4 * 4);
            rb[i] = *(const float4*)(bx + (size_t)row * KDIM + c4 * 4);
        }
    };
    auto storeLds = [&](int buf) {
#pragma unroll
        for (int i = 0; i < 4; ++i) {
            int p = tid + i * 256, row = p >> 3, c4 = p & 7;
            union { __bf16 h[4]; uint2 u; } pa, pb;
            pa.h[0] = (__bf16)ra[i].x; pa.h[1] = (__bf16)ra[i].y;
            pa.h[2] = (__bf16)ra[i].z; pa.h[3] = (__bf16)ra[i].w;
            pb.h[0] = (__bf16)rb[i].x; pb.h[1] = (__bf16)rb[i].y;
            pb.h[2] = (__bf16)rb[i].z; pb.h[3] = (__bf16)rb[i].w;
            *(uint2*)&As[buf][row * FLDT + c4 * 4] = pa.u;
            *(uint2*)&Bs[buf][row * FLDT + c4 * 4] = pb.u;
        }
    };

    const int KT = KDIM / FBK;
    loadRegs(0); storeLds(0); __syncthreads();

    for (int kt = 0; kt < KT; ++kt) {
        const int cur = kt & 1;
        if (kt + 1 < KT) loadRegs(kt + 1);

        FragBF a[2], b[4];
#pragma unroll
        for (int mi = 0; mi < 2; ++mi) {
            int row = wm * 32 + mi * 16 + r16;
            const __bf16* base = &As[cur][row * FLDT + kh * 8];
            a[mi].q[0] = *(const uint4*)(base);
            a[mi].q[1] = *(const uint4*)(base + 16);
        }
#pragma unroll
        for (int ni = 0; ni < 4; ++ni) {
            int row = wn * 64 + ni * 16 + r16;
            const __bf16* base = &Bs[cur][row * FLDT + kh * 8];
            b[ni].q[0] = *(const uint4*)(base);
            b[ni].q[1] = *(const uint4*)(base + 16);
        }
#pragma unroll
        for (int mi = 0; mi < 2; ++mi)
#pragma unroll
            for (int ni = 0; ni < 4; ++ni)
                acc[mi][ni] = __builtin_amdgcn_wmma_f32_16x16x32_bf16(
                    false, a[mi].v, false, b[ni].v,
                    (short)0, acc[mi][ni], false, false);

        if (kt + 1 < KT) { storeLds(cur ^ 1); __syncthreads(); }
    }

#pragma unroll
    for (int mi = 0; mi < 2; ++mi) {
#pragma unroll
        for (int ni = 0; ni < 4; ++ni) {
            int n = blockN + wn * 64 + ni * 16 + r16;
            float bv = bias[n];
#pragma unroll
            for (int r = 0; r < 8; ++r) {
                int m = blockM + wm * 32 + mi * 16 + kh * 8 + r;
                float h = acc[mi][ni][r] + bv;
                Out[(size_t)m * NDIM + n] = gelu_tanh(h);
            }
        }
    }
}

// =====================================================================
// LayerNorm (in place over d_out rows)
// =====================================================================
__global__ __launch_bounds__(256)
void layernorm_inplace_kernel(float* __restrict__ Out,
                              const float* __restrict__ a2,
                              const float* __restrict__ b2)
{
    __shared__ float rs[8], rs2[8];
    const int tid = threadIdx.x;
    float* g = Out + (size_t)blockIdx.x * NDIM;

    float v[16];
    float s = 0.f, s2 = 0.f;
#pragma unroll
    for (int i = 0; i < 16; ++i) {
        v[i] = g[i * 256 + tid];
        s  += v[i];
        s2 += v[i] * v[i];
    }
#pragma unroll
    for (int off = 16; off > 0; off >>= 1) {
        s  += __shfl_xor(s,  off, 32);
        s2 += __shfl_xor(s2, off, 32);
    }
    if ((tid & 31) == 0) { rs[tid >> 5] = s; rs2[tid >> 5] = s2; }
    __syncthreads();
    s = 0.f; s2 = 0.f;
#pragma unroll
    for (int w = 0; w < 8; ++w) { s += rs[w]; s2 += rs2[w]; }

    const float mean = s * (1.0f / NDIM);
    const float var  = s2 * (1.0f / NDIM) - mean * mean;
    const float inv  = rsqrtf(var + 1e-12f);
#pragma unroll
    for (int i = 0; i < 16; ++i) {
        int h = i * 256 + tid;
        g[h] = (v[i] - mean) * inv * a2[h] + b2[h];
    }
}

extern "C" void kernel_launch(void* const* d_in, const int* in_sizes, int n_in,
                              void* d_out, int out_size, void* d_ws, size_t ws_size,
                              hipStream_t stream) {
    (void)in_sizes; (void)n_in; (void)out_size;
    const float* x  = (const float*)d_in[0];
    const float* W  = (const float*)d_in[1];
    const float* b  = (const float*)d_in[2];
    const float* a2 = (const float*)d_in[3];
    const float* b2 = (const float*)d_in[4];
    float* out = (float*)d_out;

    const size_t nX = (size_t)MDIM * KDIM;           // 67.1M
    const size_t nW = (size_t)NDIM * KDIM;           // 16.8M
    const size_t need = (nX + nW) * sizeof(__bf16);  // 160 MB

    dim3 grid(NDIM / BN, MDIM / BM);   // 32 x 128: N fastest -> W stays L2-resident

    if (ws_size >= need) {
        __bf16* xb = (__bf16*)d_ws;
        __bf16* wb = xb + nX;
        cvt_f32_to_bf16_kernel<<<(unsigned)(nX / 4 / 256), 256, 0, stream>>>(x, xb);
        cvt_f32_to_bf16_kernel<<<(unsigned)(nW / 4 / 256), 256, 0, stream>>>(W, wb);
        gemm_async_kernel<<<grid, 256, 0, stream>>>(xb, wb, b, out);
    } else {
        gemm_fused_kernel<<<grid, 256, 0, stream>>>(x, W, b, out);
    }
    layernorm_inplace_kernel<<<MDIM, 256, 0, stream>>>(out, a2, b2);
}
